// SkeletonNNCLR_31327491457412
// MI455X (gfx1250) — matrix operand order repaired and checked by reading, not compiled, not run on hardware
//
#include <hip/hip_runtime.h>
#include <math.h>

#define B_SZ 512
#define D_SZ 128
#define K_SZ 131072
#define NCHUNK 64
#define KC (K_SZ / NCHUNK)   /* 2048 columns per chunk */
#define SC 64                /* columns staged per TDM superstep */
#define NS (KC / SC)         /* 32 supersteps per chunk */
#define TS (SC / 16)         /* 4 16-col tiles per superstep */
#define TEMP_F 0.07f

typedef __attribute__((ext_vector_type(16))) __bf16 v16bf;
typedef __attribute__((ext_vector_type(8)))  float  v8f;
typedef __attribute__((ext_vector_type(4)))  unsigned int u32x4;
typedef __attribute__((ext_vector_type(8)))  int i32x8;
typedef __attribute__((ext_vector_type(4)))  int i32x4;

union Frag { v16bf v; uint4 q[2]; };

__device__ __forceinline__ unsigned short f2bf(float f) {
    unsigned u = __float_as_uint(f);
    u += 0x7FFFu + ((u >> 16) & 1u);   // round-to-nearest-even
    return (unsigned short)(u >> 16);
}

// ---------------------------------------------------------------------------
// TDM: DMA a [SC rows x 128 elems] bf16 tile of qnT into LDS (contiguous).
// D# per CDNA5 ISA; 2D tensor -> trailing groups 0.
// 6-arg builtin form (clang-23 / therock-10.0 headers).
// ---------------------------------------------------------------------------
__device__ __forceinline__ void tdm_load_tile(unsigned lds_off, const unsigned short* gptr) {
    unsigned long long ga = (unsigned long long)(size_t)gptr;
    u32x4 g0 = {
        1u,                                                   // count=1, user desc
        lds_off,                                              // lds_addr (bytes)
        (unsigned)(ga & 0xFFFFFFFFull),                       // global_addr[31:0]
        (unsigned)((ga >> 32) & 0x1FFFFFFull) | (2u << 30)    // addr[56:32] | type=2
    };
    i32x8 g1 = {
        (int)(1u << 16),                                      // data_size=1 (2 bytes)
        (int)((D_SZ & 0xFFFF) << 16),                         // tensor_dim0 lo16
        (int)(((D_SZ >> 16) & 0xFFFF) | ((K_SZ & 0xFFFF) << 16)),   // tdim0 hi | tdim1 lo
        (int)((((unsigned)K_SZ >> 16) & 0xFFFF) | ((D_SZ & 0xFFFF) << 16)), // tdim1 hi | tile_dim0
        (int)(SC & 0xFFFF),                                   // tile_dim1 ; tile_dim2=0
        (int)D_SZ,                                            // tensor_dim0_stride lo32
        0, 0                                                  // stride hi / dim1_stride (unused 2D)
    };
    i32x4 gz4 = { 0, 0, 0, 0 };
    i32x8 gz8 = { 0, 0, 0, 0, 0, 0, 0, 0 };
    __builtin_amdgcn_tensor_load_to_lds(g0, g1, gz4, gz4, gz8, 0);
}

// ---------------------------------------------------------------------------
// Normalize the four [B,D] row-major inputs; emit bf16 copies of z0/z1.
// grid: (B, 4) x 128 threads
// ---------------------------------------------------------------------------
__global__ void nk_norm_rows(const float* __restrict__ pr0, const float* __restrict__ pr1,
                             const float* __restrict__ pj0, const float* __restrict__ pj1,
                             float* p0n, float* p1n, float* z0n, float* z1n,
                             unsigned short* z0b, unsigned short* z1b) {
    __shared__ float sh[128];
    int row = blockIdx.x, a = blockIdx.y, t = threadIdx.x;
    const float* src = (a == 0) ? pr0 : (a == 1) ? pr1 : (a == 2) ? pj0 : pj1;
    float v = src[row * D_SZ + t];
    sh[t] = v * v;
    __syncthreads();
    for (int s = 64; s > 0; s >>= 1) { if (t < s) sh[t] += sh[t + s]; __syncthreads(); }
    float o = v * (1.0f / sqrtf(sh[0]));
    float* dst = (a == 0) ? p0n : (a == 1) ? p1n : (a == 2) ? z0n : z1n;
    dst[row * D_SZ + t] = o;
    if (a == 2)      z0b[row * D_SZ + t] = f2bf(o);
    else if (a == 3) z1b[row * D_SZ + t] = f2bf(o);
}

// ---------------------------------------------------------------------------
// Column-normalize queue [D,K]; write f32 queue_n into d_out's new_queue slot
// and a transposed bf16 copy qnT[K,D] (WMMA B operand layout).
// grid: K/256 x 256
// ---------------------------------------------------------------------------
__global__ void nk_norm_queue(const float* __restrict__ q, float* qn, unsigned short* qnT) {
    int k = blockIdx.x * 256 + threadIdx.x;
    float s = 0.f;
    for (int d = 0; d < D_SZ; ++d) { float v = q[(size_t)d * K_SZ + k]; s += v * v; }
    float inv = 1.0f / sqrtf(s);
    for (int d = 0; d < D_SZ; ++d) {
        float v = q[(size_t)d * K_SZ + k] * inv;
        qn[(size_t)d * K_SZ + k] = v;
        qnT[(size_t)k * D_SZ + d] = f2bf(v);
    }
}

// ---------------------------------------------------------------------------
// sim = z @ queue_n with fused mask+argmax.
// grid: (NCHUNK, B/64, 2) x 128 threads (4 waves; wave = one 16-row band).
// Queue tiles staged global->LDS by the Tensor Data Mover, double buffered:
// wave 0 issues TDM for superstep s+1, waits TENSORcnt for s, barrier, and all
// waves run WMMAs from LDS (ds_load_b128 fragments) while the next DMA flies.
// ---------------------------------------------------------------------------
__global__ void nk_sim_argmax(const unsigned short* __restrict__ z0b,
                              const unsigned short* __restrict__ z1b,
                              const int* __restrict__ queue_id, const int* __restrict__ ids,
                              const unsigned short* __restrict__ qnT,
                              float* pmax, int* pidx) {
    __shared__ unsigned short lbuf[2][SC * D_SZ];   // 2 x 16KB

    int z     = blockIdx.z;
    const unsigned short* zb = z ? z1b : z0b;
    int wv    = threadIdx.x >> 5;
    int lane  = threadIdx.x & 31;
    int half  = lane >> 4;
    int l16   = lane & 15;
    int rb    = (blockIdx.y * 4 + wv) * 16;
    int chunk = blockIdx.x;
    int k0    = chunk * KC;

    // A fragments (16x32 bf16 per ISA layout), resident for the whole chunk
    Frag a[4];
    const uint4* zrow = (const uint4*)(zb + (size_t)(rb + l16) * D_SZ);
#pragma unroll
    for (int t = 0; t < 4; ++t) {
        a[t].q[0] = zrow[t * 4 + half];
        a[t].q[1] = zrow[t * 4 + 2 + half];
    }
    int myIds[8];
#pragma unroll
    for (int r = 0; r < 8; ++r) myIds[r] = ids[rb + half * 8 + r];

    float best[8]; int bidx[8];
#pragma unroll
    for (int r = 0; r < 8; ++r) { best[r] = -2.0f; bidx[r] = 0; }

    // prologue: DMA superstep 0
    if (wv == 0) tdm_load_tile((unsigned)(size_t)&lbuf[0][0],
                               qnT + (size_t)k0 * D_SZ);

    for (int s = 0; s < NS; ++s) {
        __syncthreads();   // all waves done reading buffer (s+1)&1
        if (wv == 0) {
            if (s + 1 < NS) {
                tdm_load_tile((unsigned)(size_t)&lbuf[(s + 1) & 1][0],
                              qnT + (size_t)(k0 + (s + 1) * SC) * D_SZ);
                __builtin_amdgcn_s_wait_tensorcnt(1);   // superstep s landed
            } else {
                __builtin_amdgcn_s_wait_tensorcnt(0);
            }
        }
        __syncthreads();   // buffer s&1 ready for everyone
        const unsigned short* lb = &lbuf[s & 1][0];

        // batch the mask ids for this superstep
        int qids[TS];
#pragma unroll
        for (int c = 0; c < TS; ++c) qids[c] = queue_id[k0 + s * SC + c * 16 + l16];

#pragma unroll
        for (int c = 0; c < TS; ++c) {
            int col = k0 + s * SC + c * 16 + l16;
            const uint4* qrow = (const uint4*)(lb + (size_t)(c * 16 + l16) * D_SZ);
            v8f acc;
#pragma unroll
            for (int i = 0; i < 8; ++i) acc[i] = 0.0f;
#pragma unroll
            for (int t = 0; t < 4; ++t) {
                Frag b;
                b.q[0] = qrow[t * 4 + half];
                b.q[1] = qrow[t * 4 + 2 + half];
                acc = __builtin_amdgcn_wmma_f32_16x16x32_bf16(
                    false, a[t].v, false, b.v, (short)0, acc, false, false);
            }
#pragma unroll
            for (int r = 0; r < 8; ++r) {
                float v = (qids[c] == myIds[r]) ? -1.0f : acc[r];
                if (v > best[r]) { best[r] = v; bidx[r] = col; }
            }
        }
    }

    // reduce across the 16-lane half-wave (xor 1..8 stays within halves)
#pragma unroll
    for (int off = 1; off < 16; off <<= 1) {
#pragma unroll
        for (int r = 0; r < 8; ++r) {
            float ov = __shfl_xor(best[r], off, 32);
            int   oi = __shfl_xor(bidx[r], off, 32);
            if (ov > best[r] || (ov == best[r] && oi < bidx[r])) { best[r] = ov; bidx[r] = oi; }
        }
    }
    if (l16 == 0) {
#pragma unroll
        for (int r = 0; r < 8; ++r) {
            int row = rb + half * 8 + r;
            size_t o = ((size_t)z * B_SZ + row) * NCHUNK + chunk;
            pmax[o] = best[r];
            pidx[o] = bidx[r];
        }
    }
}

// grid: 2 x 512
__global__ void nk_argmax_reduce(const float* __restrict__ pmax, const int* __restrict__ pidx,
                                 int* idxout) {
    int s = blockIdx.x, row = threadIdx.x;
    float best = -3.f; int bi = 0;
    for (int c = 0; c < NCHUNK; ++c) {
        size_t o = ((size_t)s * B_SZ + row) * NCHUNK + c;
        float v = pmax[o]; int i = pidx[o];
        if (v > best || (v == best && i < bi)) { best = v; bi = i; }
    }
    idxout[s * B_SZ + row] = bi;
}

// grid: (B, 2) x 128 — must run BEFORE the epilogue overwrites queue columns
__global__ void nk_gather(const float* __restrict__ qn, const int* __restrict__ idx,
                          float* nn0, float* nn1) {
    int row = blockIdx.x, s = blockIdx.y, d = threadIdx.x;
    int i = idx[s * B_SZ + row];
    float v = qn[(size_t)d * K_SZ + i];
    (s ? nn1 : nn0)[row * D_SZ + d] = v;
}

// grid: (B, 2) x 128.  s=0: loss0 = nnclr(p0, nn1); s=1: loss1 = nnclr(p1, nn0)
__global__ void nk_loss(const float* __restrict__ nn0, const float* __restrict__ nn1,
                        const float* __restrict__ p0n, const float* __restrict__ p1n,
                        float* terms) {
    __shared__ float nnsh[128];
    __shared__ float sh[128];
    __shared__ float diagv;
    int i = blockIdx.x, s = blockIdx.y, t = threadIdx.x;
    const float* nn = (s == 0) ? nn1 : nn0;
    const float* p  = (s == 0) ? p0n : p1n;
    nnsh[t] = nn[i * D_SZ + t];
    __syncthreads();
    float l[4]; float mx = -1e30f;
#pragma unroll
    for (int jj = 0; jj < 4; ++jj) {
        int j = jj * 128 + t;
        float dot = 0.f;
        for (int d = 0; d < D_SZ; ++d) dot += nnsh[d] * p[j * D_SZ + d];
        float lg = dot / TEMP_F;
        l[jj] = lg;
        if (lg > mx) mx = lg;
        if (j == i) diagv = lg;
    }
    sh[t] = mx; __syncthreads();
    for (int r = 64; r > 0; r >>= 1) { if (t < r) sh[t] = fmaxf(sh[t], sh[t + r]); __syncthreads(); }
    float M = sh[0]; __syncthreads();
    float e = 0.f;
#pragma unroll
    for (int jj = 0; jj < 4; ++jj) e += expf(l[jj] - M);
    sh[t] = e; __syncthreads();
    for (int r = 64; r > 0; r >>= 1) { if (t < r) sh[t] += sh[t + r]; __syncthreads(); }
    if (t == 0) terms[s * B_SZ + i] = M + logf(sh[0]) - diagv;
}

// 1 x 256, fixed-order tree => deterministic
__global__ void nk_loss_final(const float* __restrict__ terms, float* out) {
    __shared__ float sh[256];
    int t = threadIdx.x;
    sh[t] = terms[t] + terms[t + 256] + terms[t + 512] + terms[t + 768];
    __syncthreads();
    for (int r = 128; r > 0; r >>= 1) { if (t < r) sh[t] += sh[t + r]; __syncthreads(); }
    if (t == 0) out[0] = sh[0] * (1.0f / 1024.0f);
}

// grid: K/256 x 256 — overwrite first B queue columns with z0^T, write ids/ptr
__global__ void nk_epilogue(const int* __restrict__ queue_id, const int* __restrict__ ids,
                            const float* __restrict__ z0n,
                            float* qn_out, float* id_out, float* ptr_out) {
    int t = blockIdx.x * 256 + threadIdx.x;
    id_out[t] = (float)((t < B_SZ) ? ids[t] : queue_id[t]);
    if (t < B_SZ * D_SZ) {
        int d = t >> 9, b = t & 511;
        qn_out[(size_t)d * K_SZ + b] = z0n[b * D_SZ + d];
    }
    if (t == 0) ptr_out[0] = (float)((0 + B_SZ) % K_SZ);
}

extern "C" void kernel_launch(void* const* d_in, const int* in_sizes, int n_in,
                              void* d_out, int out_size, void* d_ws, size_t ws_size,
                              hipStream_t stream) {
    (void)in_sizes; (void)n_in; (void)out_size; (void)ws_size;
    const float* predict0 = (const float*)d_in[0];
    const float* predict1 = (const float*)d_in[1];
    const float* project0 = (const float*)d_in[2];
    const float* project1 = (const float*)d_in[3];
    const float* queue    = (const float*)d_in[4];
    const int*   queue_id = (const int*)d_in[5];
    const int*   ids      = (const int*)d_in[6];

    float* out      = (float*)d_out;
    float* loss_out = out;                                   // [1]
    float* qn_out   = out + 1;                               // [D*K]
    float* id_out   = out + 1 + (size_t)D_SZ * K_SZ;         // [K]
    float* ptr_out  = id_out + K_SZ;                         // [1]

    char* w = (char*)d_ws;
    float* p0n  = (float*)w; w += (size_t)B_SZ * D_SZ * 4;
    float* p1n  = (float*)w; w += (size_t)B_SZ * D_SZ * 4;
    float* z0n  = (float*)w; w += (size_t)B_SZ * D_SZ * 4;
    float* z1n  = (float*)w; w += (size_t)B_SZ * D_SZ * 4;
    float* nn0  = (float*)w; w += (size_t)B_SZ * D_SZ * 4;
    float* nn1  = (float*)w; w += (size_t)B_SZ * D_SZ * 4;
    float* pmax = (float*)w; w += (size_t)2 * B_SZ * NCHUNK * 4;
    float* terms= (float*)w; w += 1024 * 4;
    int*   pidx = (int*)w;   w += (size_t)2 * B_SZ * NCHUNK * 4;
    int*   idxb = (int*)w;   w += 2 * B_SZ * 4;
    unsigned short* z0b = (unsigned short*)w; w += (size_t)B_SZ * D_SZ * 2;
    unsigned short* z1b = (unsigned short*)w; w += (size_t)B_SZ * D_SZ * 2;
    unsigned short* qnT = (unsigned short*)w; w += (size_t)K_SZ * D_SZ * 2;

    nk_norm_rows<<<dim3(B_SZ, 4), 128, 0, stream>>>(predict0, predict1, project0, project1,
                                                    p0n, p1n, z0n, z1n, z0b, z1b);
    nk_norm_queue<<<K_SZ / 256, 256, 0, stream>>>(queue, qn_out, qnT);
    nk_sim_argmax<<<dim3(NCHUNK, B_SZ / 64, 2), 128, 0, stream>>>(z0b, z1b, queue_id, ids,
                                                                  qnT, pmax, pidx);
    nk_argmax_reduce<<<2, B_SZ, 0, stream>>>(pmax, pidx, idxb);
    nk_gather<<<dim3(B_SZ, 2), D_SZ, 0, stream>>>(qn_out, idxb, nn0, nn1);
    nk_loss<<<dim3(B_SZ, 2), 128, 0, stream>>>(nn0, nn1, p0n, p1n, terms);
    nk_loss_final<<<1, 256, 0, stream>>>(terms, loss_out);
    nk_epilogue<<<K_SZ / 256, 256, 0, stream>>>(queue_id, ids, z0n, qn_out, id_out, ptr_out);
}
